// update_v_55387898250018
// MI455X (gfx1250) — matrix operand order repaired and checked by reading, not compiled, not run on hardware
//
#include <hip/hip_runtime.h>

#define HDIM  128
#define NNODE 50000
#define NEDGE 600000
#define RDIM  6
#define BDIM  8

#ifndef USE_TDM
#define USE_TDM 1
#endif

typedef __bf16 bf16_t;
typedef bf16_t v16bf __attribute__((ext_vector_type(16)));
typedef bf16_t v8bf  __attribute__((ext_vector_type(8)));
typedef float  v8f   __attribute__((ext_vector_type(8)));
typedef float  v4f   __attribute__((ext_vector_type(4)));
typedef unsigned int u32x4 __attribute__((ext_vector_type(4)));
typedef int i32x4 __attribute__((ext_vector_type(4)));
typedef int i32x8 __attribute__((ext_vector_type(8)));

static __device__ __forceinline__ int imin(int a, int b) { return a < b ? a : b; }

static __device__ __forceinline__ float silu_f(float x) {
  return x / (1.0f + __expf(-x));
}

static __device__ __forceinline__ float fatomic_add(float* p, float v) {
  return __hip_atomic_fetch_add(p, v, __ATOMIC_RELAXED, __HIP_MEMORY_SCOPE_AGENT);
}

// Pack two float8 vectors into one v16bf via packed cvt (no v_mov_b16 fixups).
static __device__ __forceinline__ v16bf cvt2bf16(v8f lo, v8f hi) {
  v8bf blo = __builtin_convertvector(lo, v8bf);
  v8bf bhi = __builtin_convertvector(hi, v8bf);
  return __builtin_shufflevector(blo, bhi, 0, 1, 2, 3, 4, 5, 6, 7, 8, 9, 10, 11,
                                 12, 13, 14, 15);
}

// ---- WMMA operand loaders (wave32, 16x16x32 bf16) ------------------------
// A (16x32 bf16): lane holds row M = lane&15.
//   lanes 0-15:  halves 0..7 = K 0..7,   halves 8..15 = K 16..23
//   lanes 16-31: halves 0..7 = K 8..15,  halves 8..15 = K 24..31
static __device__ __forceinline__ void load_a_f32(v16bf a[4], const float* row, int lane) {
#pragma unroll
  for (int kb = 0; kb < 4; ++kb) {
    const float* p = row + kb * 32 + ((lane & 16) ? 8 : 0);
    v8f lo = *(const v8f*)p;
    v8f hi = *(const v8f*)(p + 16);
    a[kb] = cvt2bf16(lo, hi);
  }
}

// Full 16x128 = X(16xH) @ W^T (W row-major [out][in], ld=H), f32 accumulate.
// B (32x16): lane holds column N = lane&15; same K layout as A:
// lanes 0-15 K 0..7,16..23; lanes 16-31 K 8..15,24..31 (two 16-byte reads).
static __device__ __forceinline__ void gemm_full(const v16bf a[4], const bf16_t* W,
                                                 v8f acc[8], int lane) {
#pragma unroll
  for (int nt = 0; nt < 8; ++nt)
#pragma unroll
    for (int q = 0; q < 8; ++q) acc[nt][q] = 0.0f;
#pragma unroll
  for (int kb = 0; kb < 4; ++kb) {
#pragma unroll
    for (int nt = 0; nt < 8; ++nt) {
      const bf16_t* p = W + (size_t)(nt * 16 + (lane & 15)) * HDIM + kb * 32 +
                        ((lane & 16) ? 8 : 0);
      v8bf blo = *(const v8bf*)p;
      v8bf bhi = *(const v8bf*)(p + 16);
      v16bf b = __builtin_shufflevector(blo, bhi, 0, 1, 2, 3, 4, 5, 6, 7, 8, 9, 10, 11,
                                        12, 13, 14, 15);
      acc[nt] = __builtin_amdgcn_wmma_f32_16x16x32_bf16(
          false, a[kb], false, b, (short)0, acc[nt], false, false);
    }
  }
}

// ---- weight fp32 -> bf16 staging -----------------------------------------
struct CvtEnt { const float* s; bf16_t* d; int n; };
struct CvtPack { CvtEnt e[11]; };

__global__ __launch_bounds__(256) void k_cvt(CvtPack p) {
  CvtEnt en = p.e[blockIdx.y];
  for (int idx = blockIdx.x * blockDim.x + threadIdx.x; idx < en.n;
       idx += gridDim.x * blockDim.x)
    en.d[idx] = (bf16_t)en.s[idx];
}

// ---- K1: v_up_acc[i[e]] += e2[e] -----------------------------------------
__global__ __launch_bounds__(256) void k_scatter_e2(const float* e2, const int* idx_i,
                                                    float* v_up_acc) {
  int gid = blockIdx.x * blockDim.x + threadIdx.x;   // NEDGE*32 threads
  int e   = gid >> 5;
  if (e >= NEDGE) return;
  int h4 = (gid & 31) << 2;
  v4f src = *(const v4f*)(e2 + (size_t)e * HDIM + h4);
  float* d = v_up_acc + (size_t)idx_i[e] * HDIM + h4;
#pragma unroll
  for (int q = 0; q < 4; ++q) fatomic_add(d + q, src[q]);
}

// ---- K2: x_i = silu(v@Wi+bi) (f32), x_j = silu(v@Wj+bj) (bf16) -----------
__global__ __launch_bounds__(256) void k_node_proj(const float* v,
                                                   const bf16_t* Wi_b, const float* bi,
                                                   const bf16_t* Wj_b, const float* bj,
                                                   float* x_i, bf16_t* x_j_b) {
  int lane = threadIdx.x & 31, wave = threadIdx.x >> 5;
  int m0 = (blockIdx.x * 8 + wave) * 16;
  int arow = imin(m0 + (lane & 15), NNODE - 1);
  v16bf a[4];
  load_a_f32(a, v + (size_t)arow * HDIM, lane);
  v8f acc[8];

  gemm_full(a, Wi_b, acc, lane);
#pragma unroll
  for (int nt = 0; nt < 8; ++nt) {
    int col = nt * 16 + (lane & 15);
    float bv = bi[col];
#pragma unroll
    for (int r = 0; r < 8; ++r) {
      int rr = r + ((lane & 16) ? 8 : 0);
      int row = m0 + rr;
      if (row < NNODE) x_i[(size_t)row * HDIM + col] = silu_f(acc[nt][r] + bv);
    }
  }

  gemm_full(a, Wj_b, acc, lane);
#pragma unroll
  for (int nt = 0; nt < 8; ++nt) {
    int col = nt * 16 + (lane & 15);
    float bv = bj[col];
#pragma unroll
    for (int r = 0; r < 8; ++r) {
      int rr = r + ((lane & 16) ? 8 : 0);
      int row = m0 + rr;
      if (row < NNODE) x_j_b[(size_t)row * HDIM + col] = (bf16_t)silu_f(acc[nt][r] + bv);
    }
  }
}

// ---- K3: edge GEMM: silu((x_j[j]*rbf) @ Wd^T + bd), scatter on j ---------
// Wd (bf16, 32KB) is staged once per block into LDS by the Tensor Data Mover
// (one wave issues tensor_load_to_lds, waits TENSORcnt, block barriers).
__global__ __launch_bounds__(128) void k_edge(const float* rbf0, const float* Wr1,
                                              const float* Wr2, const int* idx_j,
                                              const bf16_t* x_j_b, const bf16_t* Wd_b,
                                              const float* bd, float* xj_acc) {
  __shared__ __align__(128) bf16_t Wd_lds[HDIM * HDIM];
  int lane = threadIdx.x & 31, wave = threadIdx.x >> 5;

#if USE_TDM && __has_builtin(__builtin_amdgcn_tensor_load_to_lds)
  if (wave == 0) {
    unsigned long long ga = (unsigned long long)(uintptr_t)Wd_b;
    unsigned int lds_off = (unsigned int)(uintptr_t)(void*)Wd_lds;
    // D# group0: count=1, lds_addr, global_addr[56:0], type=2 ("image")
    u32x4 g0;
    g0[0] = 1u;
    g0[1] = lds_off;
    g0[2] = (unsigned int)ga;
    g0[3] = 0x80000000u | (unsigned int)((ga >> 32) & 0x1FFFFFFull);
    // D# group1: data_size=1 (2B); tensor_dim0 = tile_dim0 = 16384 elems,
    // tensor_dim1 = tile_dim1 = 1, tensor_dim0_stride = 16384.
    i32x8 g1;
    g1[0] = 0x00010000;                 // data_size=1 at bits[17:16]
    g1[1] = (int)(16384u << 16);        // tensor_dim0 low16 at bits[63:48]
    g1[2] = 0x00010000;                 // tensor_dim1=1 at bits[95:80]
    g1[3] = (int)(16384u << 16);        // tile_dim0=16384 at bits[127:112]
    g1[4] = 1;                          // tile_dim1=1 at bits[143:128]
    g1[5] = 16384;                      // tensor_dim0_stride low32
    g1[6] = 0;
    g1[7] = 0;
    i32x4 z4;
    z4[0] = 0; z4[1] = 0; z4[2] = 0; z4[3] = 0;
    i32x8 z8;
#pragma unroll
    for (int q = 0; q < 8; ++q) z8[q] = 0;
    // 6-arg form (clang-23 / therock headers): groups 0..3 + extra + cpol
    __builtin_amdgcn_tensor_load_to_lds(g0, g1, z4, z4, z8, 0);
    __builtin_amdgcn_s_wait_tensorcnt(0);
  }
  __syncthreads();
#else
  for (int t = threadIdx.x; t < (HDIM * HDIM) / 8; t += blockDim.x)
    ((v8bf*)Wd_lds)[t] = ((const v8bf*)Wd_b)[t];
  __syncthreads();
#endif

  int e0 = (blockIdx.x * 4 + wave) * 16;          // E % 64 == 0: no guards
  int erow = e0 + (lane & 15);
  int jj = idx_j[erow];
  const bf16_t* xrow = x_j_b + (size_t)jj * HDIM;
  const float* rb = rbf0 + (size_t)erow * RDIM;

  // tiny rbf MLP: t = Wr1 @ rbf0[e] (8x6), then rbf_h = Wr2[h] . t
  float t8[BDIM];
#pragma unroll
  for (int b = 0; b < BDIM; ++b) {
    float s = 0.0f;
#pragma unroll
    for (int r = 0; r < RDIM; ++r) s += Wr1[b * RDIM + r] * rb[r];
    t8[b] = s;
  }

  v16bf a[4];
#pragma unroll
  for (int kb = 0; kb < 4; ++kb) {
    int kbase = kb * 32 + ((lane & 16) ? 8 : 0);
    v8bf lo = *(const v8bf*)(xrow + kbase);
    v8bf hi = *(const v8bf*)(xrow + kbase + 16);
    v8f flo, fhi;
#pragma unroll
    for (int t = 0; t < 8; ++t) {
      int h0 = kbase + t, h1 = kbase + 16 + t;
      float s0 = 0.0f, s1 = 0.0f;
#pragma unroll
      for (int b = 0; b < BDIM; ++b) {
        s0 += Wr2[h0 * BDIM + b] * t8[b];
        s1 += Wr2[h1 * BDIM + b] * t8[b];
      }
      flo[t] = (float)lo[t] * s0;
      fhi[t] = (float)hi[t] * s1;
    }
    a[kb] = cvt2bf16(flo, fhi);
  }

  v8f acc[8];
  gemm_full(a, Wd_lds, acc, lane);   // B tiles read from LDS (ds_load_b128)

#pragma unroll
  for (int nt = 0; nt < 8; ++nt) {
    int col = nt * 16 + (lane & 15);
    float bv = bd[col];
#pragma unroll
    for (int r = 0; r < 8; ++r) {
      int rr = r + ((lane & 16) ? 8 : 0);
      int jj2 = idx_j[e0 + rr];
      fatomic_add(xj_acc + (size_t)jj2 * HDIM + col, silu_f(acc[nt][r] + bv));
    }
  }
}

// ---- K4a: Wu, (Wgu), Wc, residual(Wb1,Wb2) -> v2_mid ---------------------
__global__ __launch_bounds__(256) void k_node_mid(const float* v_up_acc, const float* xj_acc,
                                                  const float* x_i,
                                                  const bf16_t* Wgu_b, const float* bgu,
                                                  const bf16_t* Wu_b, const float* bu,
                                                  const bf16_t* Wc_b, const float* bc,
                                                  const bf16_t* Wb1_b, const float* bb1,
                                                  const bf16_t* Wb2_b, const float* bb2,
                                                  float* v2_mid) {
  __shared__ float xch[8][16][HDIM];               // 64 KB, one tile per wave
  int lane = threadIdx.x & 31, wave = threadIdx.x >> 5;
  int m0 = (blockIdx.x * 8 + wave) * 16;
  int arow = imin(m0 + (lane & 15), NNODE - 1);
  float(*tile)[HDIM] = xch[wave];
  v16bf a[4];
  v8f acc[8], keep[8];

  // stage 1: v2 = silu(xj_acc@Wu+bu) + x_i  -> tile
  load_a_f32(a, xj_acc + (size_t)arow * HDIM, lane);
  gemm_full(a, Wu_b, acc, lane);
#pragma unroll
  for (int nt = 0; nt < 8; ++nt) {
    int col = nt * 16 + (lane & 15);
    float bv = bu[col];
#pragma unroll
    for (int r = 0; r < 8; ++r) {
      int rr = r + ((lane & 16) ? 8 : 0);
      int rowc = imin(m0 + rr, NNODE - 1);
      tile[rr][col] = silu_f(acc[nt][r] + bv) + x_i[(size_t)rowc * HDIM + col];
    }
  }
  __syncthreads();

  // stage 2: keep = silu(v_up_acc@Wgu+bgu) + silu(v2@Wc+bc)  -> tile
  load_a_f32(a, v_up_acc + (size_t)arow * HDIM, lane);
  gemm_full(a, Wgu_b, keep, lane);
#pragma unroll
  for (int nt = 0; nt < 8; ++nt) {
    int col = nt * 16 + (lane & 15);
    float bv = bgu[col];
#pragma unroll
    for (int r = 0; r < 8; ++r) keep[nt][r] = silu_f(keep[nt][r] + bv);
  }
  load_a_f32(a, tile[lane & 15], lane);
  gemm_full(a, Wc_b, acc, lane);
  __syncthreads();
#pragma unroll
  for (int nt = 0; nt < 8; ++nt) {
    int col = nt * 16 + (lane & 15);
    float bv = bc[col];
#pragma unroll
    for (int r = 0; r < 8; ++r) {
      int rr = r + ((lane & 16) ? 8 : 0);
      keep[nt][r] += silu_f(acc[nt][r] + bv);
      tile[rr][col] = keep[nt][r];
    }
  }
  __syncthreads();

  // stage 3: residual block (Wb1, Wb2)
  load_a_f32(a, tile[lane & 15], lane);
  gemm_full(a, Wb1_b, acc, lane);
  __syncthreads();
#pragma unroll
  for (int nt = 0; nt < 8; ++nt) {
    int col = nt * 16 + (lane & 15);
    float bv = bb1[col];
#pragma unroll
    for (int r = 0; r < 8; ++r) {
      int rr = r + ((lane & 16) ? 8 : 0);
      tile[rr][col] = silu_f(acc[nt][r] + bv);
    }
  }
  __syncthreads();
  load_a_f32(a, tile[lane & 15], lane);
  gemm_full(a, Wb2_b, acc, lane);
#pragma unroll
  for (int nt = 0; nt < 8; ++nt) {
    int col = nt * 16 + (lane & 15);
    float bv = bb2[col];
#pragma unroll
    for (int r = 0; r < 8; ++r) {
      int rr = r + ((lane & 16) ? 8 : 0);
      int row = m0 + rr;
      float val = keep[nt][r] + silu_f(acc[nt][r] + bv);
      if (row < NNODE) v2_mid[(size_t)row * HDIM + col] = val;
    }
  }
}

// ---- K4b: Wl (+v skip), 2x residual(Wa1,Wa2), W1 projection --------------
__global__ __launch_bounds__(256) void k_node_out(const float* v2_mid, const float* v,
                                                  const bf16_t* Wl_b, const float* bl,
                                                  const bf16_t* Wa1_b, const float* ba1,
                                                  const bf16_t* Wa2_b, const float* ba2,
                                                  const float* W1, float* out_v2,
                                                  float* out_v1) {
  __shared__ float xch[8][16][HDIM];
  int lane = threadIdx.x & 31, wave = threadIdx.x >> 5;
  int m0 = (blockIdx.x * 8 + wave) * 16;
  int arow = imin(m0 + (lane & 15), NNODE - 1);
  float(*tile)[HDIM] = xch[wave];
  v16bf a[4];
  v8f acc[8], keep[8];

  // v2 = silu(v2_mid@Wl+bl) + v
  load_a_f32(a, v2_mid + (size_t)arow * HDIM, lane);
  gemm_full(a, Wl_b, acc, lane);
#pragma unroll
  for (int nt = 0; nt < 8; ++nt) {
    int col = nt * 16 + (lane & 15);
    float bv = bl[col];
#pragma unroll
    for (int r = 0; r < 8; ++r) {
      int rr = r + ((lane & 16) ? 8 : 0);
      int rowc = imin(m0 + rr, NNODE - 1);
      keep[nt][r] = silu_f(acc[nt][r] + bv) + v[(size_t)rowc * HDIM + col];
      tile[rr][col] = keep[nt][r];
    }
  }
  __syncthreads();

#pragma unroll 1
  for (int k = 0; k < 2; ++k) {
    load_a_f32(a, tile[lane & 15], lane);
    gemm_full(a, Wa1_b + (size_t)k * HDIM * HDIM, acc, lane);
    __syncthreads();
#pragma unroll
    for (int nt = 0; nt < 8; ++nt) {
      int col = nt * 16 + (lane & 15);
      float bv = ba1[k * HDIM + col];
#pragma unroll
      for (int r = 0; r < 8; ++r) {
        int rr = r + ((lane & 16) ? 8 : 0);
        tile[rr][col] = silu_f(acc[nt][r] + bv);
      }
    }
    __syncthreads();
    load_a_f32(a, tile[lane & 15], lane);
    gemm_full(a, Wa2_b + (size_t)k * HDIM * HDIM, acc, lane);
    __syncthreads();
#pragma unroll
    for (int nt = 0; nt < 8; ++nt) {
      int col = nt * 16 + (lane & 15);
      float bv = ba2[k * HDIM + col];
#pragma unroll
      for (int r = 0; r < 8; ++r) {
        int rr = r + ((lane & 16) ? 8 : 0);
        keep[nt][r] += silu_f(acc[nt][r] + bv);
        tile[rr][col] = keep[nt][r];
      }
    }
    __syncthreads();
  }

  // store v2, and v1 = v2 @ W1^T (H->1) via half-wave shuffle reduction
  float part[8];
#pragma unroll
  for (int r = 0; r < 8; ++r) part[r] = 0.0f;
#pragma unroll
  for (int nt = 0; nt < 8; ++nt) {
    int col = nt * 16 + (lane & 15);
    float w = W1[col];
#pragma unroll
    for (int r = 0; r < 8; ++r) {
      int rr = r + ((lane & 16) ? 8 : 0);
      int row = m0 + rr;
      if (row < NNODE) out_v2[(size_t)row * HDIM + col] = keep[nt][r];
      part[r] += keep[nt][r] * w;
    }
  }
#pragma unroll
  for (int r = 0; r < 8; ++r) {
    float s = part[r];
    s += __shfl_xor(s, 1, 32);
    s += __shfl_xor(s, 2, 32);
    s += __shfl_xor(s, 4, 32);
    s += __shfl_xor(s, 8, 32);
    int rr = r + ((lane & 16) ? 8 : 0);
    int row = m0 + rr;
    if ((lane & 15) == 0 && row < NNODE) out_v1[row] = s;
  }
}

extern "C" void kernel_launch(void* const* d_in, const int* in_sizes, int n_in,
                              void* d_out, int out_size, void* d_ws, size_t ws_size,
                              hipStream_t stream) {
  const float* v    = (const float*)d_in[0];
  const float* rbf0 = (const float*)d_in[1];
  const float* e2   = (const float*)d_in[2];
  const float* Wgu  = (const float*)d_in[3];
  const float* bgu  = (const float*)d_in[4];
  const float* Wi   = (const float*)d_in[5];
  const float* bi   = (const float*)d_in[6];
  const float* Wj   = (const float*)d_in[7];
  const float* bj   = (const float*)d_in[8];
  const float* Wr1  = (const float*)d_in[9];
  const float* Wr2  = (const float*)d_in[10];
  const float* Wd   = (const float*)d_in[11];
  const float* bd   = (const float*)d_in[12];
  const float* Wu   = (const float*)d_in[13];
  const float* bu   = (const float*)d_in[14];
  const float* Wc   = (const float*)d_in[15];
  const float* bc   = (const float*)d_in[16];
  const float* Wl   = (const float*)d_in[17];
  const float* bl   = (const float*)d_in[18];
  const float* Wb1  = (const float*)d_in[19];
  const float* bb1  = (const float*)d_in[20];
  const float* Wb2  = (const float*)d_in[21];
  const float* bb2  = (const float*)d_in[22];
  const float* Wa1  = (const float*)d_in[23];
  const float* ba1  = (const float*)d_in[24];
  const float* Wa2  = (const float*)d_in[25];
  const float* ba2  = (const float*)d_in[26];
  const float* W1   = (const float*)d_in[27];
  const int* idx_i  = (const int*)d_in[28];
  const int* idx_j  = (const int*)d_in[29];

  const size_t NH = (size_t)NNODE * HDIM;
  const int HH = HDIM * HDIM;

  float* ws = (float*)d_ws;
  float* v_up_acc = ws;
  float* xj_acc   = ws + NH;
  float* x_i      = ws + 2 * NH;
  float* v2_mid   = ws + 3 * NH;
  bf16_t* bws     = (bf16_t*)(ws + 4 * NH);
  bf16_t* x_j_b   = bws;
  bf16_t* Wslab   = bws + NH;
  bf16_t* Wgu_b = Wslab + (size_t)0 * HH;
  bf16_t* Wi_b  = Wslab + (size_t)1 * HH;
  bf16_t* Wj_b  = Wslab + (size_t)2 * HH;
  bf16_t* Wd_b  = Wslab + (size_t)3 * HH;
  bf16_t* Wu_b  = Wslab + (size_t)4 * HH;
  bf16_t* Wc_b  = Wslab + (size_t)5 * HH;
  bf16_t* Wb1_b = Wslab + (size_t)6 * HH;
  bf16_t* Wb2_b = Wslab + (size_t)7 * HH;
  bf16_t* Wl_b  = Wslab + (size_t)8 * HH;
  bf16_t* Wa1_b = Wslab + (size_t)9 * HH;   // 2 matrices
  bf16_t* Wa2_b = Wslab + (size_t)11 * HH;  // 2 matrices

  // zero the two scatter accumulators
  (void)hipMemsetAsync(v_up_acc, 0, 2 * NH * sizeof(float), stream);

  CvtPack p;
  p.e[0]  = {Wgu, Wgu_b, HH};
  p.e[1]  = {Wi,  Wi_b,  HH};
  p.e[2]  = {Wj,  Wj_b,  HH};
  p.e[3]  = {Wd,  Wd_b,  HH};
  p.e[4]  = {Wu,  Wu_b,  HH};
  p.e[5]  = {Wc,  Wc_b,  HH};
  p.e[6]  = {Wb1, Wb1_b, HH};
  p.e[7]  = {Wb2, Wb2_b, HH};
  p.e[8]  = {Wl,  Wl_b,  HH};
  p.e[9]  = {Wa1, Wa1_b, 2 * HH};
  p.e[10] = {Wa2, Wa2_b, 2 * HH};
  k_cvt<<<dim3(32, 11), 256, 0, stream>>>(p);

  k_scatter_e2<<<(NEDGE * 32) / 256, 256, 0, stream>>>(e2, idx_i, v_up_acc);

  int node_blocks = (NNODE + 127) / 128;
  k_node_proj<<<node_blocks, 256, 0, stream>>>(v, Wi_b, bi, Wj_b, bj, x_i, x_j_b);

  k_edge<<<NEDGE / 64, 128, 0, stream>>>(rbf0, Wr1, Wr2, idx_j, x_j_b, Wd_b, bd, xj_acc);

  k_node_mid<<<node_blocks, 256, 0, stream>>>(v_up_acc, xj_acc, x_i, Wgu_b, bgu, Wu_b, bu,
                                              Wc_b, bc, Wb1_b, bb1, Wb2_b, bb2, v2_mid);

  float* out_v2 = (float*)d_out;
  float* out_v1 = out_v2 + NH;
  k_node_out<<<node_blocks, 256, 0, stream>>>(v2_mid, v, Wl_b, bl, Wa1_b, ba1, Wa2_b, ba2,
                                              W1, out_v2, out_v1);
}
